// BaseLayerWithLoRACPU_72980084293847
// MI455X (gfx1250) — matrix operand order repaired and checked by reading, not compiled, not run on hardware
//
#include <hip/hip_runtime.h>

typedef __attribute__((ext_vector_type(16))) __bf16 v16bf;
typedef __attribute__((ext_vector_type(8)))  float  v8f;
typedef __attribute__((ext_vector_type(4)))  float  f32x4;   // native vector for NT loads

#define DIN    8192
#define DOUT   8192
#define MTOT   128
#define RANK   16
#define KCHUNK 64
#define NROWS  32          // N columns (W rows) per workgroup
#define NSUB   2           // NROWS/16
#define LDS_STRIDE 72      // bf16 elems/row: 64 data + 8 pad (144 B, 16B aligned, conflict-free)

union F32U { float f; unsigned u; };
union V16U { v16bf v; unsigned u[8]; };

// Pack the two bf16 "hi" halves of two fp32 bit patterns in ONE v_perm_b32:
// result = [u1[31:16] : u0[31:16]]
__device__ __forceinline__ unsigned pack_bf16_hi(unsigned u0, unsigned u1) {
  return __builtin_amdgcn_perm(u1, u0, 0x07060302u);
}
__device__ __forceinline__ float trunc_hi_f32(unsigned u) {
  F32U c; c.u = u & 0xFFFF0000u; return c.f;
}

// fp32[2] -> packed bf16 hi pair + packed bf16 lo (truncated residual) pair.
// (clang lowers the residual to a single v_fma_mix_f32_bf16.)
__device__ __forceinline__ void split_pair(float f0, float f1, unsigned& hp, unsigned& lp) {
  F32U a0, a1; a0.f = f0; a1.f = f1;
  hp = pack_bf16_hi(a0.u, a1.u);
  F32U r0, r1;
  r0.f = f0 - trunc_hi_f32(a0.u);
  r1.f = f1 - trunc_hi_f32(a1.u);
  lp = pack_bf16_hi(r0.u, r1.u);
}

// ---------------- Kernel 0: one-shot split of x into bf16 hi/lo planes --------------
// x is 4 MB; splitting it once (0.4 us) removes ALL per-WG conversion VALU from the
// 256 workgroups of the main GEMM. xhi/xlo stay L2-resident (192 MB L2).
__global__ __launch_bounds__(256) void split_x_kernel(const float* __restrict__ x,
                                                      __bf16* __restrict__ xhi,
                                                      __bf16* __restrict__ xlo) {
  const size_t i4 = (size_t)(blockIdx.x * 256 + threadIdx.x) * 4;
  const float4 v = *(const float4*)(x + i4);
  uint2 hp, lp;
  split_pair(v.x, v.y, hp.x, lp.x);
  split_pair(v.z, v.w, hp.y, lp.y);
  *(uint2*)(xhi + i4) = hp;
  *(uint2*)(xlo + i4) = lp;
}

// ---------------- Kernel 1: T[m][r] = sum_k x[m][k] * A[r][k]  (128x16, tiny) --------
__global__ __launch_bounds__(256) void lora_xa_kernel(const float* __restrict__ x,
                                                      const float* __restrict__ A,
                                                      float* __restrict__ T) {
  __shared__ float red[256];
  const int m = blockIdx.x;
  const int t = threadIdx.x;
  const int r = t >> 4;
  const int l = t & 15;
  const float* xr = x + (size_t)m * DIN;
  const float* Ar = A + (size_t)r * DIN;
  float acc = 0.f;
  for (int k = l; k < DIN; k += 16) acc += xr[k] * Ar[k];
  red[t] = acc;
  __syncthreads();
  if (t < 16) {
    float s = 0.f;
#pragma unroll
    for (int i = 0; i < 16; ++i) s += red[t * 16 + i];
    T[m * RANK + t] = s;
  }
}

// ---------------- Kernel 2: out = x @ W^T + b + T @ loraB^T  (3xBF16 WMMA) ----------
// 4 waves / WG. Wave tile = 2 M-tiles x 2 N-subtiles (12 WMMAs per fragment-load set).
// WG tile = M128 x N32; grid 256 WGs -> W read once from HBM (non-temporal).
// Hot loop per wave/K-step: 8 global b128 (A hi/lo) + 8 ds b128 (B hi/lo) + 12 WMMA.
__global__ __launch_bounds__(128) void fused_lora_gemm(const __bf16* __restrict__ xhi,
                                                       const __bf16* __restrict__ xlo,
                                                       const float* __restrict__ W,
                                                       const float* __restrict__ bias,
                                                       const float* __restrict__ loraB,
                                                       const float* __restrict__ T,
                                                       float* __restrict__ out) {
  __shared__ __bf16 ldsHi[2][NROWS * LDS_STRIDE];   // 9 KB
  __shared__ __bf16 ldsLo[2][NROWS * LDS_STRIDE];   // 9 KB
  __shared__ float  ldsT[MTOT * RANK];              // 8 KB

  const int tid  = threadIdx.x;
  const int wave = tid >> 5;        // 0..3
  const int lane = tid & 31;
  const int half = lane >> 4;
  const int l15  = lane & 15;
  const int n0   = blockIdx.x * NROWS;
  const int m0   = wave * 32;       // two 16-row M tiles per wave

  for (int i = tid; i < MTOT * RANK; i += 128) ldsT[i] = T[i];

  v8f acc[2][NSUB] = {};

  // A layout (16-bit 16x32): lanes 0-15 row m hold K {0..7, 16..23}; lanes 16-31 +8.
  const size_t arow0 = (size_t)(m0 + l15) * DIN + half * 8;   // element offset
  const size_t arow1 = arow0 + (size_t)16 * DIN;

  auto load_chunk = [&](int kc, int buf) {
#pragma unroll
    for (int q = 0; q < 4; ++q) {
      const int idx = q * 128 + tid;       // 512 float4 groups = 32 rows x 64 K, coalesced
      const int row = idx >> 4;
      const int c4  = idx & 15;
      // W is consumed exactly once: stream it non-temporally so it never evicts
      // the L2-resident xhi/xlo planes.
      const f32x4 w4 = __builtin_nontemporal_load(
          (const f32x4*)(W + (size_t)(n0 + row) * (size_t)DIN + kc + c4 * 4));
      uint2 hp, lp;
      split_pair(w4.x, w4.y, hp.x, lp.x);
      split_pair(w4.z, w4.w, hp.y, lp.y);
      const int off = row * LDS_STRIDE + c4 * 4;
      *reinterpret_cast<uint2*>(&ldsHi[buf][off]) = hp;   // ds_store_b64
      *reinterpret_cast<uint2*>(&ldsLo[buf][off]) = lp;
    }
  };

  load_chunk(0, 0);
  __syncthreads();

  const int NCHUNK = DIN / KCHUNK;
  for (int c = 0; c < NCHUNK; ++c) {
    const int buf = c & 1;
    if (c + 1 < NCHUNK) load_chunk((c + 1) * KCHUNK, buf ^ 1);  // double-buffer prefetch

    const int kc = c * KCHUNK;
#pragma unroll
    for (int ks = 0; ks < KCHUNK; ks += 32) {
      // A fragments: straight b128 loads from the pre-split planes (no VALU).
      V16U ah0, al0, ah1, al1;
      const size_t a0 = arow0 + kc + ks;
      const size_t a1 = arow1 + kc + ks;
      *(uint4*)&ah0.u[0] = *(const uint4*)(xhi + a0);
      *(uint4*)&ah0.u[4] = *(const uint4*)(xhi + a0 + 16);
      *(uint4*)&al0.u[0] = *(const uint4*)(xlo + a0);
      *(uint4*)&al0.u[4] = *(const uint4*)(xlo + a0 + 16);
      *(uint4*)&ah1.u[0] = *(const uint4*)(xhi + a1);
      *(uint4*)&ah1.u[4] = *(const uint4*)(xhi + a1 + 16);
      *(uint4*)&al1.u[0] = *(const uint4*)(xlo + a1);
      *(uint4*)&al1.u[4] = *(const uint4*)(xlo + a1 + 16);
#pragma unroll
      for (int sub = 0; sub < NSUB; ++sub) {
        // B (32x16): lane<16 holds col n=l15, K {0..7, 16..23}; lane>=16 +8.
        const int boff = (sub * 16 + l15) * LDS_STRIDE + ks + half * 8;
        V16U bh, bl;
        *(uint4*)&bh.u[0] = *reinterpret_cast<const uint4*>(&ldsHi[buf][boff]);
        *(uint4*)&bh.u[4] = *reinterpret_cast<const uint4*>(&ldsHi[buf][boff + 16]);
        *(uint4*)&bl.u[0] = *reinterpret_cast<const uint4*>(&ldsLo[buf][boff]);
        *(uint4*)&bl.u[4] = *reinterpret_cast<const uint4*>(&ldsLo[buf][boff + 16]);
        acc[0][sub] = __builtin_amdgcn_wmma_f32_16x16x32_bf16(false, ah0.v, false, bh.v,
                                                              (short)0, acc[0][sub], false, false);
        acc[0][sub] = __builtin_amdgcn_wmma_f32_16x16x32_bf16(false, ah0.v, false, bl.v,
                                                              (short)0, acc[0][sub], false, false);
        acc[0][sub] = __builtin_amdgcn_wmma_f32_16x16x32_bf16(false, al0.v, false, bh.v,
                                                              (short)0, acc[0][sub], false, false);
        acc[1][sub] = __builtin_amdgcn_wmma_f32_16x16x32_bf16(false, ah1.v, false, bh.v,
                                                              (short)0, acc[1][sub], false, false);
        acc[1][sub] = __builtin_amdgcn_wmma_f32_16x16x32_bf16(false, ah1.v, false, bl.v,
                                                              (short)0, acc[1][sub], false, false);
        acc[1][sub] = __builtin_amdgcn_wmma_f32_16x16x32_bf16(false, al1.v, false, bh.v,
                                                              (short)0, acc[1][sub], false, false);
      }
    }
    __syncthreads();  // protects next iteration's LDS writes (2-deep pipeline)
  }

  // Epilogue (fp32): + bias + (x @ A^T) @ B^T.
  // C/D layout: lane<16 -> n=l15, M=v; lane>=16 -> n=l15, M=v+8.
#pragma unroll
  for (int sub = 0; sub < NSUB; ++sub) {
    const int n = n0 + sub * 16 + l15;
    const float bv = bias[n];
    float lbv[16];
    const float* lbp = loraB + (size_t)n * RANK;
    *(float4*)&lbv[0]  = *(const float4*)(lbp + 0);
    *(float4*)&lbv[4]  = *(const float4*)(lbp + 4);
    *(float4*)&lbv[8]  = *(const float4*)(lbp + 8);
    *(float4*)&lbv[12] = *(const float4*)(lbp + 12);
#pragma unroll
    for (int mt = 0; mt < 2; ++mt) {
#pragma unroll
      for (int v = 0; v < 8; ++v) {
        const int m = m0 + mt * 16 + v + half * 8;
        const float* Tm = &ldsT[m * RANK];
        float s = bv;
#pragma unroll
        for (int r = 0; r < RANK; ++r) s += Tm[r] * lbv[r];
        __builtin_nontemporal_store(acc[mt][sub][v] + s, out + (size_t)m * DOUT + n);
      }
    }
  }
}

extern "C" void kernel_launch(void* const* d_in, const int* in_sizes, int n_in,
                              void* d_out, int out_size, void* d_ws, size_t ws_size,
                              hipStream_t stream) {
  const float* x  = (const float*)d_in[0];   // (8,16,8192)
  const float* W  = (const float*)d_in[1];   // (8192,8192)
  const float* b  = (const float*)d_in[2];   // (8192,)
  const float* lA = (const float*)d_in[3];   // (16,8192)
  const float* lB = (const float*)d_in[4];   // (8192,16)
  float* out = (float*)d_out;                // (8,16,8192)

  // Workspace layout: T (8 KB) | xhi (2 MB) | xlo (2 MB)
  float*  T   = (float*)d_ws;
  __bf16* xhi = (__bf16*)((char*)d_ws + 8192);
  __bf16* xlo = xhi + (size_t)MTOT * DIN;

  hipLaunchKernelGGL(split_x_kernel, dim3((MTOT * DIN / 4) / 256), dim3(256), 0, stream,
                     x, xhi, xlo);
  hipLaunchKernelGGL(lora_xa_kernel, dim3(MTOT), dim3(256), 0, stream, x, lA, T);
  hipLaunchKernelGGL(fused_lora_gemm, dim3(DOUT / NROWS), dim3(128), 0, stream,
                     xhi, xlo, W, b, lB, T, out);
}
